// HyperAttention_89610197663834
// MI455X (gfx1250) — matrix operand order repaired and checked by reading, not compiled, not run on hardware
//
#include <hip/hip_runtime.h>
#include <hip/hip_bf16.h>

// HyperAttention for MI455X (gfx1250, wave32).
//  B=4, T=2048, D=64, N_HASHES=8, SAMPLE=256 (128 heavy + 128 light), SCALE=1/8
//
// Pipeline:
//  k1: LSH hash codes (8-bit) for q and k rows               -> ws
//  k2: k -> f16 (WMMA operand precision, halves gather bytes) -> ws
//  k3: fused per-query wave: counting top-k over hash overlap, WMMA scores
//      (v_wmma_f32_16x16x32_f16), register softmax, coalesced VALU output GEMV.
// Workspace: 8KB + 8KB + 1MB = ~1.07MB required.

#define B_   4
#define T_   2048
#define D_   64
#define NH_  8
#define SS_  256
#define HK_  128
#define LK_  128
#define SCALE_ 0.125f

typedef __attribute__((ext_vector_type(16))) _Float16 v16h;
typedef __attribute__((ext_vector_type(2)))  _Float16 v2h;
typedef __attribute__((ext_vector_type(8)))  float    v8f;

// ---------------------------------------------------------------------------
// Kernel 1: 8-bit LSH codes. One thread per row; R (64x8) staged in LDS.
// ---------------------------------------------------------------------------
__global__ __launch_bounds__(256) void hyper_hash_kernel(
    const float* __restrict__ q, const float* __restrict__ k,
    const float* __restrict__ R,
    unsigned char* __restrict__ cq, unsigned char* __restrict__ ck) {
  __shared__ float Rl[D_ * NH_];
  const int tid = threadIdx.x;
  Rl[tid]       = R[tid];
  Rl[tid + 256] = R[tid + 256];
  __syncthreads();

  const int r = blockIdx.x * 256 + tid;  // row in [0, B*T)
  const float4* q4 = (const float4*)(q + (size_t)r * D_);
  const float4* k4 = (const float4*)(k + (size_t)r * D_);
  float aq[NH_], ak[NH_];
#pragma unroll
  for (int h = 0; h < NH_; ++h) { aq[h] = 0.f; ak[h] = 0.f; }
#pragma unroll
  for (int dv = 0; dv < 16; ++dv) {
    float4 xq = q4[dv], xk = k4[dv];
#pragma unroll
    for (int h = 0; h < NH_; ++h) {
      aq[h] += xq.x * Rl[(4 * dv + 0) * NH_ + h] + xq.y * Rl[(4 * dv + 1) * NH_ + h]
             + xq.z * Rl[(4 * dv + 2) * NH_ + h] + xq.w * Rl[(4 * dv + 3) * NH_ + h];
      ak[h] += xk.x * Rl[(4 * dv + 0) * NH_ + h] + xk.y * Rl[(4 * dv + 1) * NH_ + h]
             + xk.z * Rl[(4 * dv + 2) * NH_ + h] + xk.w * Rl[(4 * dv + 3) * NH_ + h];
    }
  }
  unsigned bq = 0, bk = 0;
#pragma unroll
  for (int h = 0; h < NH_; ++h) {
    bq |= (aq[h] >= 0.f ? 1u : 0u) << h;
    bk |= (ak[h] >= 0.f ? 1u : 0u) << h;
  }
  cq[r] = (unsigned char)bq;
  ck[r] = (unsigned char)bk;
}

// ---------------------------------------------------------------------------
// Kernel 2: k -> f16, fully coalesced (2 elems / thread).
// ---------------------------------------------------------------------------
__global__ __launch_bounds__(256) void hyper_cvt_kernel(
    const float* __restrict__ k, _Float16* __restrict__ kf) {
  size_t i = ((size_t)blockIdx.x * 256 + threadIdx.x) * 2;
  float2 x = *(const float2*)(k + i);
  v2h o;
  o[0] = (_Float16)x.x;
  o[1] = (_Float16)x.y;
  *(v2h*)(kf + i) = o;
}

// ---------------------------------------------------------------------------
// Kernel 3: fused selection + WMMA attention. One wave32 per query,
// 8 waves (8 queries) per 256-thread block.
// ---------------------------------------------------------------------------
__global__ __launch_bounds__(256) void hyper_attn_kernel(
    const float* __restrict__ q, const float* __restrict__ v,
    const _Float16* __restrict__ kf,
    const unsigned char* __restrict__ cq, const unsigned char* __restrict__ ck,
    const int* __restrict__ light, float* __restrict__ out) {
  __shared__ int      cand[8][SS_];
  __shared__ float    wl[8][SS_];
  __shared__ _Float16 qh[8][D_];

  const int lane = threadIdx.x & 31;
  const int wv   = threadIdx.x >> 5;
  const int qi   = blockIdx.x * 8 + wv;  // query row in [0, B*T)
  const int b    = qi / T_;

  // ---------- Phase A: exact top-128 by hash overlap (counting select) ------
  // Lane owns keys [lane*64, lane*64+64). overlap in [0,8]: 9-bin histogram
  // packed into one u64 (7-bit fields), wave-reduced as two 12-bit-field u64s.
  const unsigned qc4 = (unsigned)cq[qi] * 0x01010101u;
  const uint4* ckp = (const uint4*)(ck + (size_t)b * T_ + lane * 64);
  unsigned ovp[8];
#pragma unroll
  for (int i = 0; i < 8; ++i) ovp[i] = 0u;
  unsigned long long hist = 0ull;
#pragma unroll
  for (int wvec = 0; wvec < 4; ++wvec) {
    uint4 cw = ckp[wvec];
    unsigned ww[4] = {cw.x, cw.y, cw.z, cw.w};
#pragma unroll
    for (int wi = 0; wi < 4; ++wi) {
      unsigned x = ww[wi] ^ qc4;
#pragma unroll
      for (int by = 0; by < 4; ++by) {
        unsigned ov = 8u - (unsigned)__popc((x >> (8 * by)) & 0xFFu);
        int kk = wvec * 16 + wi * 4 + by;
        ovp[kk >> 3] |= ov << ((kk & 7) * 4);
        hist += 1ull << (ov * 7);
      }
    }
  }
  // Wave-wide histogram.
  unsigned long long lo = 0ull, hi = 0ull;
#pragma unroll
  for (int vv = 0; vv < 9; ++vv) {
    unsigned long long c = (hist >> (7 * vv)) & 0x7Full;
    if (vv < 5) lo += c << (12 * vv);
    else        hi += c << (12 * (vv - 5));
  }
#pragma unroll
  for (int o = 16; o > 0; o >>= 1) {
    lo += __shfl_xor(lo, o);
    hi += __shfl_xor(hi, o);
  }
  unsigned g[9];
#pragma unroll
  for (int vv = 0; vv < 9; ++vv)
    g[vv] = (vv < 5) ? (unsigned)((lo >> (12 * vv)) & 0xFFFull)
                     : (unsigned)((hi >> (12 * (vv - 5))) & 0xFFFull);
  // Threshold: vstar = max v with count(ov >= v) >= 128 (top_k tie rule:
  // all ov > vstar taken, ov == vstar filled in ascending index order).
  int vstar = 0; unsigned ngt = 0, suf = 0;
#pragma unroll
  for (int vv = 8; vv >= 0; --vv) {
    unsigned nsuf = suf + g[vv];
    if (nsuf >= HK_ && suf < HK_) { vstar = vv; ngt = suf; }
    suf = nsuf;
  }
  const unsigned need_eq = HK_ - ngt;
  // Per-lane gt/eq counts + exclusive wave prefix (packed in one u32).
  unsigned mygt = 0, myeq = 0;
#pragma unroll
  for (int vv = 0; vv < 9; ++vv) {
    unsigned c = (unsigned)((hist >> (7 * vv)) & 0x7Full);
    if (vv > vstar)  mygt += c;
    if (vv == vstar) myeq  = c;
  }
  unsigned packed = mygt | (myeq << 16);
  unsigned incl = packed;
#pragma unroll
  for (int o = 1; o < 32; o <<= 1) {
    unsigned tsh = __shfl_up(incl, o);
    if (lane >= o) incl += tsh;
  }
  const unsigned excl = incl - packed;
  const unsigned pgt = excl & 0xFFFFu, peq = excl >> 16;
  unsigned cg = 0, ce = 0;
  const int j0 = lane * 64;
#pragma unroll
  for (int kk = 0; kk < 64; ++kk) {
    unsigned ov = (ovp[kk >> 3] >> ((kk & 7) * 4)) & 0xFu;
    if ((int)ov > vstar) {
      cand[wv][pgt + cg] = j0 + kk; ++cg;
    } else if ((int)ov == vstar) {
      unsigned rk = peq + ce; ++ce;
      if (rk < need_eq) cand[wv][ngt + rk] = j0 + kk;
    }
  }
  // Light candidates + query -> f16 staged for WMMA A fragment.
  const int* lp = light + (size_t)qi * LK_;
#pragma unroll
  for (int i = 0; i < LK_ / 32; ++i) cand[wv][HK_ + lane + 32 * i] = lp[lane + 32 * i];
  {
    float2 qq = *(const float2*)(q + (size_t)qi * D_ + 2 * lane);
    qh[wv][2 * lane]     = (_Float16)qq.x;
    qh[wv][2 * lane + 1] = (_Float16)qq.y;
  }
  __syncthreads();

  // ---------- Phase B: scores via v_wmma_f32_16x16x32_f16 -------------------
  // A (16x32 f16): query replicated across the 16 rows, contiguous K order
  // matching B: lane<16 halves -> K {0..15}, lane>=16 -> K {16..31}.
  v16h a0, a1;
  {
    const int kb = (lane >= 16) ? 16 : 0;
#pragma unroll
    for (int h = 0; h < 16; ++h) {
      int K = h + kb;
      a0[h] = qh[wv][K];
      a1[h] = qh[wv][32 + K];
    }
  }
  // B (32x16 f16): lane = column (candidate), halves = contiguous K chunk ->
  // each lane does one contiguous 32B gather from its candidate's f16 K-row.
  const int col  = lane & 15;
  const int koff = (lane >= 16) ? 16 : 0;
  float sc[16];
#pragma unroll
  for (int t16 = 0; t16 < 16; ++t16) {
    int row = cand[wv][t16 * 16 + col];
    const _Float16* kp = kf + ((size_t)b * T_ + row) * D_;
    v16h b0 = *(const v16h*)(kp + koff);
    v16h b1 = *(const v16h*)(kp + 32 + koff);
    v8f acc = {};
    acc = __builtin_amdgcn_wmma_f32_16x16x32_f16(false, a0, false, b0,
                                                 (short)0, acc, false, false);
    acc = __builtin_amdgcn_wmma_f32_16x16x32_f16(false, a1, false, b1,
                                                 (short)0, acc, false, false);
    sc[t16] = acc[0] * SCALE_;  // rows replicated: acc[0] = score of this col
  }
  // ---------- softmax (registers + 16-lane xor reduction) -------------------
  float m = -1e30f;
#pragma unroll
  for (int t16 = 0; t16 < 16; ++t16) m = fmaxf(m, sc[t16]);
#pragma unroll
  for (int o = 8; o > 0; o >>= 1) m = fmaxf(m, __shfl_xor(m, o));
  float s = 0.f;
#pragma unroll
  for (int t16 = 0; t16 < 16; ++t16) { sc[t16] = __expf(sc[t16] - m); s += sc[t16]; }
#pragma unroll
  for (int o = 8; o > 0; o >>= 1) s += __shfl_xor(s, o);
  const float inv = 1.0f / s;
  if (lane < 16) {
#pragma unroll
    for (int t16 = 0; t16 < 16; ++t16) wl[wv][t16 * 16 + col] = sc[t16] * inv;
  }
  __syncthreads();

  // ---------- Phase C: out = attn @ V, coalesced 256B V-row reads -----------
  float2 oacc = {0.f, 0.f};
  const float* vb = v + (size_t)b * T_ * D_;
#pragma unroll 4
  for (int j = 0; j < SS_; ++j) {
    int row = cand[wv][j];
    float w = wl[wv][j];
    float2 vv2 = *(const float2*)(vb + (size_t)row * D_ + 2 * lane);
    oacc.x += w * vv2.x;
    oacc.y += w * vv2.y;
  }
  *(float2*)(out + (size_t)qi * D_ + 2 * lane) = oacc;
}

// ---------------------------------------------------------------------------
extern "C" void kernel_launch(void* const* d_in, const int* in_sizes, int n_in,
                              void* d_out, int out_size, void* d_ws, size_t ws_size,
                              hipStream_t stream) {
  const float* q     = (const float*)d_in[0];
  const float* k     = (const float*)d_in[1];
  const float* v     = (const float*)d_in[2];
  const float* R     = (const float*)d_in[3];
  const int*   light = (const int*)d_in[4];
  float*       out   = (float*)d_out;

  // Workspace layout (needs ~1.07MB): codes_q | codes_k | k_f16
  unsigned char* cq = (unsigned char*)d_ws;                        // 8192 B
  unsigned char* ck = cq + (size_t)B_ * T_;                        // 8192 B
  _Float16*      kf = (_Float16*)((char*)d_ws + 2 * (size_t)B_ * T_); // 1 MB, 16KB-aligned

  hyper_hash_kernel<<<(B_ * T_) / 256, 256, 0, stream>>>(q, k, R, cq, ck);
  hyper_cvt_kernel<<<(B_ * T_ * D_ / 2) / 256, 256, 0, stream>>>(k, kf);
  hyper_attn_kernel<<<(B_ * T_) / 8, 256, 0, stream>>>(q, v, kf, cq, ck, light, out);
}
